// ENLCA_20375324852760
// MI455X (gfx1250) — matrix operand order repaired and checked by reading, not compiled, not run on hardware
//
#include <hip/hip_runtime.h>

// Problem constants (from reference setup_inputs)
#define Nn   16
#define Cc   128
#define CR   64
#define MM   128
#define HWT  16384
#define TBLK (HWT / 64)   // 256 token tiles of 64 per batch
#define CTX_CHUNKS 8
#define CHUNK_T (HWT / CTX_CHUNKS)  // 2048

static constexpr float K_AMP = 2.449489742783178f;    // sqrt(6)
static constexpr float DN    = 0.3535533905932738f;   // 64^-0.25
static constexpr float RATIO = 0.08838834764831845f;  // 128^-0.5
static constexpr float DIAG  = 0.375f;                // 0.5 * 6 * dn^2 (||q||=sqrt(6) exactly)
static constexpr float EPSK  = 1e-4f;
static constexpr float EPSN  = 5e-5f;
static constexpr float RES   = 0.1f;

typedef __attribute__((ext_vector_type(16))) __bf16 bf16x16;
typedef __attribute__((ext_vector_type(8)))  __bf16 bf16x8;
typedef __attribute__((ext_vector_type(8)))  float  f32x8;

// ---------------- WMMA fragment helpers (CDNA5 wave32 layouts) ----------------
// A: 16x32 bf16. lane m = lane&15, hi = lane>>4.
//    elems 0..7  -> k = hi*8 + j        (8 contiguous halves)
//    elems 8..15 -> k = 16 + hi*8 + j-8 (8 contiguous halves)
__device__ __forceinline__ bf16x16 frag_a(const __bf16* base, int row0, int pitch,
                                          int kbase, int lane) {
  int m = lane & 15, hi = lane >> 4;
  const __bf16* p = base + (row0 + m) * pitch + kbase + hi * 8;
  bf16x8 lo = *(const bf16x8*)p;
  bf16x8 hh = *(const bf16x8*)(p + 16);
  bf16x16 a;
#pragma unroll
  for (int i = 0; i < 8; ++i) { a[i] = lo[i]; a[8 + i] = hh[i]; }
  return a;
}

// B: 32x16 bf16 staged in LDS as [n][k] (k contiguous). lane n = lane&15, hi = lane>>4.
//    elems 0..7 -> k = hi*8 + j; elems 8..15 -> k = 16 + hi*8 + j-8
__device__ __forceinline__ bf16x16 frag_b(const __bf16* base, int col0, int pitch,
                                          int kbase, int lane) {
  int n = lane & 15, hi = lane >> 4;
  const __bf16* p = base + (col0 + n) * pitch + kbase + hi * 8;
  bf16x8 lo = *(const bf16x8*)p;
  bf16x8 hh = *(const bf16x8*)(p + 16);
  bf16x16 b;
#pragma unroll
  for (int i = 0; i < 8; ++i) { b[i] = lo[i]; b[8 + i] = hh[i]; }
  return b;
}

__device__ __forceinline__ f32x8 wmma_bf16(bf16x16 a, bf16x16 b, f32x8 c) {
  return __builtin_amdgcn_wmma_f32_16x16x32_bf16(false, a, false, b, (short)0, c,
                                                 false, false);
}

// ---------------- Kernel 0: weights -> bf16 ----------------
__global__ void k_prep(const float* __restrict__ w1, const float* __restrict__ w2,
                       const float* __restrict__ wa, const float* __restrict__ proj,
                       __bf16* w1b, __bf16* w2b, __bf16* wab, __bf16* projb) {
  int i = blockIdx.x * 256 + threadIdx.x;
  if (i < CR * Cc) { w1b[i] = (__bf16)w1[i]; w2b[i] = (__bf16)w2[i]; }
  if (i < Cc * Cc) wab[i] = (__bf16)wa[i];
  if (i < MM * CR) projb[i] = (__bf16)proj[i];
}

// ---------------- Kernel 1: q/k GEMM + column L2-norm, store (q*dn) bf16 ----------------
// OutT[Cr][HW] = l2norm_col(W[Cr][C] @ X[C][HW] + bias) * K_AMP * DN
__global__ __launch_bounds__(256) void k_qk(const float* __restrict__ x,
                                            const __bf16* __restrict__ Wb,
                                            const float* __restrict__ bias,
                                            __bf16* __restrict__ Out) {
  const int XP = Cc + 8;  // 136
  __shared__ __align__(16) __bf16 ldsW[CR * Cc];
  __shared__ __align__(16) __bf16 ldsX[64 * (Cc + 8)];
  __shared__ float epi[CR * 65];
  __shared__ float cscale[64];
  int tid = threadIdx.x, lane = tid & 31, wave = tid >> 5;
  int n = blockIdx.y, t0 = blockIdx.x * 64;
  const float* xb = x + (size_t)n * Cc * HWT;

  for (int i = tid; i < CR * Cc; i += 256) ldsW[i] = Wb[i];
  for (int i = tid; i < 64 * Cc; i += 256) {
    int t = i & 63, c = i >> 6;
    ldsX[t * XP + c] = (__bf16)xb[(size_t)c * HWT + t0 + t];
  }
  __syncthreads();

  int m0 = (wave & 3) * 16;
  int nhalf = (wave >> 2) * 32;
#pragma unroll
  for (int sub = 0; sub < 2; ++sub) {
    int n0 = nhalf + sub * 16;
    f32x8 acc = {0, 0, 0, 0, 0, 0, 0, 0};
#pragma unroll
    for (int kc = 0; kc < Cc; kc += 32) {
      bf16x16 a = frag_a(ldsW, m0, Cc, kc, lane);
      bf16x16 b = frag_b(ldsX, n0, XP, kc, lane);
      acc = wmma_bf16(a, b, acc);
    }
    int nn = lane & 15, hh = lane >> 4;
#pragma unroll
    for (int r = 0; r < 8; ++r) {
      int m = m0 + hh * 8 + r;
      epi[m * 65 + n0 + nn] = acc[r] + bias[m];
    }
  }
  __syncthreads();

  if (tid < 64) {
    float s = 0.f;
    for (int m = 0; m < CR; ++m) { float v = epi[m * 65 + tid]; s += v * v; }
    float nrm = fmaxf(sqrtf(s), EPSN);
    cscale[tid] = (K_AMP * DN) / nrm;
  }
  __syncthreads();

  __bf16* ob = Out + (size_t)n * CR * HWT;
  for (int i = tid; i < CR * 64; i += 256) {
    int t = i & 63, m = i >> 6;
    ob[(size_t)m * HWT + t0 + t] = (__bf16)(epi[m * 65 + t] * cscale[t]);
  }
}

// ---------------- Kernel 2: v GEMM, store bf16 ----------------
// VT[C][HW] = Wa[C][C] @ X[C][HW] + ba
__global__ __launch_bounds__(256) void k_v(const float* __restrict__ x,
                                           const __bf16* __restrict__ Wb,
                                           const float* __restrict__ bias,
                                           __bf16* __restrict__ Out) {
  const int XP = Cc + 8;
  __shared__ __align__(16) __bf16 ldsW[Cc * Cc];
  __shared__ __align__(16) __bf16 ldsX[64 * (Cc + 8)];
  int tid = threadIdx.x, lane = tid & 31, wave = tid >> 5;
  int n = blockIdx.y, t0 = blockIdx.x * 64;
  const float* xb = x + (size_t)n * Cc * HWT;

  for (int i = tid; i < Cc * Cc; i += 256) ldsW[i] = Wb[i];
  for (int i = tid; i < 64 * Cc; i += 256) {
    int t = i & 63, c = i >> 6;
    ldsX[t * XP + c] = (__bf16)xb[(size_t)c * HWT + t0 + t];
  }
  __syncthreads();

  int m0 = wave * 16;
  __bf16* ob = Out + (size_t)n * Cc * HWT;
#pragma unroll
  for (int sub = 0; sub < 4; ++sub) {
    int n0 = sub * 16;
    f32x8 acc = {0, 0, 0, 0, 0, 0, 0, 0};
#pragma unroll
    for (int kc = 0; kc < Cc; kc += 32) {
      bf16x16 a = frag_a(ldsW, m0, Cc, kc, lane);
      bf16x16 b = frag_b(ldsX, n0, XP, kc, lane);
      acc = wmma_bf16(a, b, acc);
    }
    int nn = lane & 15, hh = lane >> 4;
#pragma unroll
    for (int r = 0; r < 8; ++r) {
      int e = m0 + hh * 8 + r;
      ob[(size_t)e * HWT + t0 + n0 + nn] = (__bf16)(acc[r] + bias[e]);
    }
  }
}

// ---------------- Kernel 3: qp = ratio*(exp(qd - 0.375 - colmax) + eps) ----------------
// qdT[M][HW] = proj[M][Cr] @ QT[Cr][HW]
__global__ __launch_bounds__(256) void k_qp(const __bf16* __restrict__ Qb,
                                            const __bf16* __restrict__ projb,
                                            __bf16* __restrict__ QPb) {
  const int XP = CR + 8;  // 72
  __shared__ __align__(16) __bf16 ldsW[MM * CR];
  __shared__ __align__(16) __bf16 ldsX[64 * (CR + 8)];
  __shared__ float epi[MM * 65];
  __shared__ float cmax[64];
  int tid = threadIdx.x, lane = tid & 31, wave = tid >> 5;
  int n = blockIdx.y, t0 = blockIdx.x * 64;
  const __bf16* qb = Qb + (size_t)n * CR * HWT;

  for (int i = tid; i < MM * CR; i += 256) ldsW[i] = projb[i];
  for (int i = tid; i < 64 * CR; i += 256) {
    int t = i & 63, c = i >> 6;
    ldsX[t * XP + c] = qb[(size_t)c * HWT + t0 + t];
  }
  __syncthreads();

  int m0 = wave * 16;
#pragma unroll
  for (int sub = 0; sub < 4; ++sub) {
    int n0 = sub * 16;
    f32x8 acc = {0, 0, 0, 0, 0, 0, 0, 0};
#pragma unroll
    for (int kc = 0; kc < CR; kc += 32) {
      bf16x16 a = frag_a(ldsW, m0, CR, kc, lane);
      bf16x16 b = frag_b(ldsX, n0, XP, kc, lane);
      acc = wmma_bf16(a, b, acc);
    }
    int nn = lane & 15, hh = lane >> 4;
#pragma unroll
    for (int r = 0; r < 8; ++r) epi[(m0 + hh * 8 + r) * 65 + n0 + nn] = acc[r];
  }
  __syncthreads();

  if (tid < 64) {
    float mx = -3.4e38f;
    for (int m = 0; m < MM; ++m) mx = fmaxf(mx, epi[m * 65 + tid]);
    cmax[tid] = mx;
  }
  __syncthreads();

  __bf16* ob = QPb + (size_t)n * MM * HWT;
  for (int i = tid; i < MM * 64; i += 256) {
    int t = i & 63, m = i >> 6;
    float v = RATIO * (expf(epi[m * 65 + t] - DIAG - cmax[t]) + EPSK);
    ob[(size_t)m * HWT + t0 + t] = (__bf16)v;
  }
}

// ---------------- Kernel 4: kd GEMM, per-block max partials ----------------
__global__ __launch_bounds__(256) void k_kdmax(const __bf16* __restrict__ Kb,
                                               const __bf16* __restrict__ projb,
                                               float* __restrict__ kmax_part) {
  const int XP = CR + 8;
  __shared__ __align__(16) __bf16 ldsW[MM * CR];
  __shared__ __align__(16) __bf16 ldsX[64 * (CR + 8)];
  __shared__ float red[256];
  int tid = threadIdx.x, lane = tid & 31, wave = tid >> 5;
  int n = blockIdx.y, t0 = blockIdx.x * 64;
  const __bf16* kb = Kb + (size_t)n * CR * HWT;

  for (int i = tid; i < MM * CR; i += 256) ldsW[i] = projb[i];
  for (int i = tid; i < 64 * CR; i += 256) {
    int t = i & 63, c = i >> 6;
    ldsX[t * XP + c] = kb[(size_t)c * HWT + t0 + t];
  }
  __syncthreads();

  int m0 = wave * 16;
  float lm = -3.4e38f;
#pragma unroll
  for (int sub = 0; sub < 4; ++sub) {
    int n0 = sub * 16;
    f32x8 acc = {0, 0, 0, 0, 0, 0, 0, 0};
#pragma unroll
    for (int kc = 0; kc < CR; kc += 32) {
      bf16x16 a = frag_a(ldsW, m0, CR, kc, lane);
      bf16x16 b = frag_b(ldsX, n0, XP, kc, lane);
      acc = wmma_bf16(a, b, acc);
    }
#pragma unroll
    for (int r = 0; r < 8; ++r) lm = fmaxf(lm, acc[r]);
  }
  red[tid] = lm;
  __syncthreads();
  if (tid == 0) {
    float mx = -3.4e38f;
    for (int i = 0; i < 256; ++i) mx = fmaxf(mx, red[i]);
    kmax_part[blockIdx.y * gridDim.x + blockIdx.x] = mx;
  }
}

__global__ void k_redmax(const float* __restrict__ part, float* __restrict__ kmax) {
  __shared__ float red[256];
  int tid = threadIdx.x;
  float mx = -3.4e38f;
  for (int i = tid; i < Nn * TBLK; i += 256) mx = fmaxf(mx, part[i]);
  red[tid] = mx;
  __syncthreads();
  if (tid == 0) {
    for (int i = 1; i < 256; ++i) mx = fmaxf(mx, red[i]);
    *kmax = mx;
  }
}

// ---------------- Kernel 5: kp + ksum partials ----------------
__global__ __launch_bounds__(256) void k_kp(const __bf16* __restrict__ Kb,
                                            const __bf16* __restrict__ projb,
                                            const float* __restrict__ kmax,
                                            __bf16* __restrict__ KPb,
                                            float* __restrict__ ksum_part) {
  const int XP = CR + 8;
  __shared__ __align__(16) __bf16 ldsW[MM * CR];
  __shared__ __align__(16) __bf16 ldsX[64 * (CR + 8)];
  __shared__ float epi[MM * 65];
  int tid = threadIdx.x, lane = tid & 31, wave = tid >> 5;
  int n = blockIdx.y, t0 = blockIdx.x * 64;
  const __bf16* kb = Kb + (size_t)n * CR * HWT;
  float kmx = *kmax;

  for (int i = tid; i < MM * CR; i += 256) ldsW[i] = projb[i];
  for (int i = tid; i < 64 * CR; i += 256) {
    int t = i & 63, c = i >> 6;
    ldsX[t * XP + c] = kb[(size_t)c * HWT + t0 + t];
  }
  __syncthreads();

  int m0 = wave * 16;
#pragma unroll
  for (int sub = 0; sub < 4; ++sub) {
    int n0 = sub * 16;
    f32x8 acc = {0, 0, 0, 0, 0, 0, 0, 0};
#pragma unroll
    for (int kc = 0; kc < CR; kc += 32) {
      bf16x16 a = frag_a(ldsW, m0, CR, kc, lane);
      bf16x16 b = frag_b(ldsX, n0, XP, kc, lane);
      acc = wmma_bf16(a, b, acc);
    }
    int nn = lane & 15, hh = lane >> 4;
#pragma unroll
    for (int r = 0; r < 8; ++r) epi[(m0 + hh * 8 + r) * 65 + n0 + nn] = acc[r];
  }
  __syncthreads();

  __bf16* ob = KPb + (size_t)n * MM * HWT;
  for (int i = tid; i < MM * 64; i += 256) {
    int t = i & 63, m = i >> 6;
    float v = RATIO * (expf(epi[m * 65 + t] - DIAG - kmx) + EPSK);
    epi[m * 65 + t] = v;  // keep f32 for row sums
    ob[(size_t)m * HWT + t0 + t] = (__bf16)v;
  }
  __syncthreads();

  if (tid < MM) {
    float s = 0.f;
    for (int t = 0; t < 64; ++t) s += epi[tid * 65 + t];
    ksum_part[((size_t)n * TBLK + blockIdx.x) * MM + tid] = s;
  }
}

__global__ void k_redksum(const float* __restrict__ part, float* __restrict__ ksum) {
  int idx = blockIdx.x * 256 + threadIdx.x;
  if (idx >= Nn * MM) return;
  int n = idx >> 7, m = idx & 127;
  float s = 0.f;
  for (int j = 0; j < TBLK; ++j) s += part[((size_t)n * TBLK + j) * MM + m];
  ksum[n * MM + m] = s;
}

// ---------------- Kernel 6: ctx split-K GEMM: ctx[m][e] = sum_t kp[m][t]*v[e][t] ----------------
__global__ __launch_bounds__(256) void k_ctx(const __bf16* __restrict__ KPb,
                                             const __bf16* __restrict__ VTb,
                                             float* __restrict__ ctx_part) {
  const int AP = 64 + 8;  // 72
  __shared__ __align__(16) __bf16 ldsA[MM * (64 + 8)];
  __shared__ __align__(16) __bf16 ldsB[Cc * (64 + 8)];
  int tid = threadIdx.x, lane = tid & 31, wave = tid >> 5;
  int chunk = blockIdx.x, n = blockIdx.y;
  const __bf16* ap = KPb + (size_t)n * MM * HWT;
  const __bf16* bp = VTb + (size_t)n * Cc * HWT;

  f32x8 acc[8];
#pragma unroll
  for (int s = 0; s < 8; ++s) acc[s] = (f32x8){0, 0, 0, 0, 0, 0, 0, 0};

  int m0 = wave * 16;
  for (int step = 0; step < CHUNK_T / 64; ++step) {
    int tb = chunk * CHUNK_T + step * 64;
    for (int i = tid; i < MM * 64; i += 256) {
      int t = i & 63, m = i >> 6;
      ldsA[m * AP + t] = ap[(size_t)m * HWT + tb + t];
    }
    for (int i = tid; i < Cc * 64; i += 256) {
      int t = i & 63, e = i >> 6;
      ldsB[e * AP + t] = bp[(size_t)e * HWT + tb + t];
    }
    __syncthreads();
#pragma unroll
    for (int kc = 0; kc < 64; kc += 32) {
      bf16x16 a = frag_a(ldsA, m0, AP, kc, lane);
#pragma unroll
      for (int s = 0; s < 8; ++s) {
        bf16x16 b = frag_b(ldsB, s * 16, AP, kc, lane);
        acc[s] = wmma_bf16(a, b, acc[s]);
      }
    }
    __syncthreads();
  }

  float* op = ctx_part + ((size_t)(n * CTX_CHUNKS + chunk)) * MM * Cc;
  int nn = lane & 15, hh = lane >> 4;
#pragma unroll
  for (int s = 0; s < 8; ++s)
#pragma unroll
    for (int r = 0; r < 8; ++r)
      op[(m0 + hh * 8 + r) * Cc + s * 16 + nn] = acc[s][r];
}

// reduce ctx chunks; transpose to ctxT[e][m] bf16 for the final GEMM A-operand
__global__ void k_redctx(const float* __restrict__ part, __bf16* __restrict__ ctxTb) {
  int n = blockIdx.x;
  for (int idx = threadIdx.x; idx < MM * Cc; idx += 256) {
    int e = idx >> 7, m = idx & 127;
    float s = 0.f;
    for (int c = 0; c < CTX_CHUNKS; ++c)
      s += part[((size_t)(n * CTX_CHUNKS + c) * MM + m) * Cc + e];
    ctxTb[((size_t)n * Cc + e) * MM + m] = (__bf16)s;
  }
}

// ---------------- Kernel 7: out[e][t] = 0.1 * d_inv[t] * sum_m ctxT[e][m]*qp[m][t] ----------------
__global__ __launch_bounds__(256) void k_final(const __bf16* __restrict__ ctxTb,
                                               const __bf16* __restrict__ QPb,
                                               const float* __restrict__ ksum,
                                               float* __restrict__ out) {
  const int QP = MM + 8;  // 136
  __shared__ __align__(16) __bf16 ldsW[Cc * MM];
  __shared__ __align__(16) __bf16 ldsQ[64 * (MM + 8)];
  __shared__ float ksumL[MM];
  __shared__ float dinv[64];
  int tid = threadIdx.x, lane = tid & 31, wave = tid >> 5;
  int n = blockIdx.y, t0 = blockIdx.x * 64;
  const __bf16* wb = ctxTb + (size_t)n * Cc * MM;
  const __bf16* qb = QPb + (size_t)n * MM * HWT;

  for (int i = tid; i < Cc * MM; i += 256) ldsW[i] = wb[i];
  for (int i = tid; i < 64 * MM; i += 256) {
    int t = i & 63, m = i >> 6;
    ldsQ[t * QP + m] = qb[(size_t)m * HWT + t0 + t];
  }
  if (tid < MM) ksumL[tid] = ksum[n * MM + tid];
  __syncthreads();

  if (tid < 64) {
    float s = 0.f;
    for (int m = 0; m < MM; ++m) s += (float)ldsQ[tid * QP + m] * ksumL[m];
    dinv[tid] = RES / s;
  }
  __syncthreads();

  int m0 = wave * 16;  // channel rows e
  float* ob = out + (size_t)n * Cc * HWT;
#pragma unroll
  for (int sub = 0; sub < 4; ++sub) {
    int n0 = sub * 16;
    f32x8 acc = {0, 0, 0, 0, 0, 0, 0, 0};
#pragma unroll
    for (int kc = 0; kc < MM; kc += 32) {
      bf16x16 a = frag_a(ldsW, m0, MM, kc, lane);
      bf16x16 b = frag_b(ldsQ, n0, QP, kc, lane);
      acc = wmma_bf16(a, b, acc);
    }
    int nn = lane & 15, hh = lane >> 4;
    float dv = dinv[n0 + nn];
#pragma unroll
    for (int r = 0; r < 8; ++r) {
      int e = m0 + hh * 8 + r;
      ob[(size_t)e * HWT + t0 + n0 + nn] = acc[r] * dv;
    }
  }
}

// ---------------- launcher ----------------
extern "C" void kernel_launch(void* const* d_in, const int* in_sizes, int n_in,
                              void* d_out, int out_size, void* d_ws, size_t ws_size,
                              hipStream_t stream) {
  (void)in_sizes; (void)n_in; (void)out_size; (void)ws_size;
  const float* x    = (const float*)d_in[0];
  const float* w1   = (const float*)d_in[1];
  const float* b1   = (const float*)d_in[2];
  const float* w2   = (const float*)d_in[3];
  const float* b2   = (const float*)d_in[4];
  const float* wa   = (const float*)d_in[5];
  const float* ba   = (const float*)d_in[6];
  const float* proj = (const float*)d_in[7];
  float* out = (float*)d_out;

  char* ws = (char*)d_ws;
  size_t off = 0;
  auto take = [&](size_t bytes) -> char* {
    char* p = ws + off;
    off = (off + bytes + 255) & ~(size_t)255;
    return p;
  };
  __bf16* w1b      = (__bf16*)take((size_t)CR * Cc * 2);
  __bf16* w2b      = (__bf16*)take((size_t)CR * Cc * 2);
  __bf16* wab      = (__bf16*)take((size_t)Cc * Cc * 2);
  __bf16* projb    = (__bf16*)take((size_t)MM * CR * 2);
  __bf16* QTb      = (__bf16*)take((size_t)Nn * CR * HWT * 2);
  __bf16* KTb      = (__bf16*)take((size_t)Nn * CR * HWT * 2);
  __bf16* VTb      = (__bf16*)take((size_t)Nn * Cc * HWT * 2);
  __bf16* QPb      = (__bf16*)take((size_t)Nn * MM * HWT * 2);
  __bf16* KPb      = (__bf16*)take((size_t)Nn * MM * HWT * 2);
  float* kmax_part = (float*)take((size_t)Nn * TBLK * 4);
  float* kmax      = (float*)take(256);
  float* ksum_part = (float*)take((size_t)Nn * TBLK * MM * 4);
  float* ksum      = (float*)take((size_t)Nn * MM * 4);
  float* ctx_part  = (float*)take((size_t)Nn * CTX_CHUNKS * MM * Cc * 4);
  __bf16* ctxTb    = (__bf16*)take((size_t)Nn * Cc * MM * 2);

  dim3 gTile(TBLK, Nn);

  k_prep<<<64, 256, 0, stream>>>(w1, w2, wa, proj, w1b, w2b, wab, projb);
  k_qk<<<gTile, 256, 0, stream>>>(x, w1b, b1, QTb);
  k_qk<<<gTile, 256, 0, stream>>>(x, w2b, b2, KTb);
  k_v<<<gTile, 256, 0, stream>>>(x, wab, ba, VTb);
  k_qp<<<gTile, 256, 0, stream>>>(QTb, projb, QPb);
  k_kdmax<<<gTile, 256, 0, stream>>>(KTb, projb, kmax_part);
  k_redmax<<<1, 256, 0, stream>>>(kmax_part, kmax);
  k_kp<<<gTile, 256, 0, stream>>>(KTb, projb, kmax, KPb, ksum_part);
  k_redksum<<<(Nn * MM + 255) / 256, 256, 0, stream>>>(ksum_part, ksum);
  k_ctx<<<dim3(CTX_CHUNKS, Nn), 256, 0, stream>>>(KPb, VTb, ctx_part);
  k_redctx<<<Nn, 256, 0, stream>>>(ctx_part, ctxTb);
  k_final<<<gTile, 256, 0, stream>>>(ctxTb, QPb, ksum, out);
}
